// SpikingMLP_75634374082830
// MI455X (gfx1250) — matrix operand order repaired and checked by reading, not compiled, not run on hardware
//
#include <hip/hip_runtime.h>

// ---------------------------------------------------------------------------
// Types
// ---------------------------------------------------------------------------
typedef __attribute__((ext_vector_type(16))) _Float16 v16h;
typedef __attribute__((ext_vector_type(8)))  _Float16 v8h;
typedef __attribute__((ext_vector_type(8)))  float    v8f;
typedef __attribute__((ext_vector_type(4)))  int      v4i;

// AS-qualified element typedefs so we can form addrspace-qualified pointers
typedef __attribute__((address_space(1))) v4i as1_v4i;  // global
typedef __attribute__((address_space(3))) v4i as3_v4i;  // LDS

#if defined(__HIP_DEVICE_COMPILE__) && __has_builtin(__builtin_amdgcn_global_load_async_to_lds_b128)
#define USE_ASYNC_LDS 1
#else
#define USE_ASYNC_LDS 0
#endif

// ---------------------------------------------------------------------------
// f32 -> f16 convert (grid-stride)
// ---------------------------------------------------------------------------
__global__ __launch_bounds__(256) void cvt_f32_to_f16(const float* __restrict__ in,
                                                      _Float16* __restrict__ out,
                                                      size_t n) {
  size_t i = (size_t)blockIdx.x * blockDim.x + threadIdx.x;
  size_t stride = (size_t)gridDim.x * blockDim.x;
  for (; i < n; i += stride) out[i] = (_Float16)in[i];
}

// W: [K][N] row-major f32  ->  Wt: [N][K] row-major f16  (Wt[n][k] = W[k][n])
__global__ __launch_bounds__(256) void transpose_f32_to_f16(const float* __restrict__ W,
                                                            _Float16* __restrict__ Wt,
                                                            int K, int N) {
  size_t total = (size_t)K * N;
  size_t i = (size_t)blockIdx.x * blockDim.x + threadIdx.x;
  size_t stride = (size_t)gridDim.x * blockDim.x;
  for (; i < total; i += stride) {
    int k = (int)(i / N);
    int n = (int)(i % N);
    Wt[(size_t)n * K + k] = (_Float16)W[i];
  }
}

// ---------------------------------------------------------------------------
// Tiled WMMA GEMM:  C[M,N] = A[M,K] (f16, row-major) x Bt[N,K]^T (f16) + bias[N]
// 128x128 block tile, K-step 32, 256 threads = 8 waves (4 M-waves x 2 N-waves),
// each wave computes 2x4 = 8 WMMA 16x16 tiles with v_wmma_f32_16x16x32_f16.
// ---------------------------------------------------------------------------
__global__ __launch_bounds__(256)
void gemm_f16_wmma(const _Float16* __restrict__ A, const _Float16* __restrict__ Bt,
                   const float* __restrict__ bias, float* __restrict__ C,
                   int M, int N, int K) {
  __shared__ __align__(16) _Float16 As[128 * 32];  // [m][k]
  __shared__ __align__(16) _Float16 Bs[128 * 32];  // [n][k]

  const int tid   = threadIdx.x;
  const int lane  = tid & 31;
  const int wave  = tid >> 5;
  const int waveM = wave & 3;   // M offset = 32 * waveM
  const int waveN = wave >> 2;  // N offset = 64 * waveN
  const int blockM = blockIdx.y * 128;
  const int blockN = blockIdx.x * 128;

  // staging: each thread moves 32B of A-tile and 32B of B-tile per K-step
  const int ldRow = tid >> 1;         // 0..127
  const int ldSeg = (tid & 1) * 16;   // 0 or 16 halfs

  const v8f vzero = {0.f, 0.f, 0.f, 0.f, 0.f, 0.f, 0.f, 0.f};
  v8f acc[2][4];
#pragma unroll
  for (int i = 0; i < 2; ++i)
#pragma unroll
    for (int j = 0; j < 4; ++j) acc[i][j] = vzero;

  // Lane k layout this code uses for 16-bit WMMA operands (16x16x32, contiguous, same for A and B):
  // A (16x32): lanes 0-15 hold K{0..15};       lanes 16-31 hold K{16..31}
  // B (32x16): lanes 0-15 hold K{0..15};       lanes 16-31 hold K{16..31}
  const int aKlo = (lane & 16) ? 16 : 0;
  const int bKlo = (lane & 16) ? 16 : 0;
  const int mrow = lane & 15;

  for (int k0 = 0; k0 < K; k0 += 32) {
    const _Float16* gA = A  + (size_t)(blockM + ldRow) * K + k0 + ldSeg;
    const _Float16* gB = Bt + (size_t)(blockN + ldRow) * K + k0 + ldSeg;
    _Float16* lA = &As[ldRow * 32 + ldSeg];
    _Float16* lB = &Bs[ldRow * 32 + ldSeg];

#if USE_ASYNC_LDS
    // gfx1250 async global->LDS DMA; tracked by ASYNCcnt
    __builtin_amdgcn_global_load_async_to_lds_b128((as1_v4i*)(void*)gA,       (as3_v4i*)(void*)lA,       0, 0);
    __builtin_amdgcn_global_load_async_to_lds_b128((as1_v4i*)(void*)(gA + 8), (as3_v4i*)(void*)(lA + 8), 0, 0);
    __builtin_amdgcn_global_load_async_to_lds_b128((as1_v4i*)(void*)gB,       (as3_v4i*)(void*)lB,       0, 0);
    __builtin_amdgcn_global_load_async_to_lds_b128((as1_v4i*)(void*)(gB + 8), (as3_v4i*)(void*)(lB + 8), 0, 0);
#if __has_builtin(__builtin_amdgcn_s_wait_asynccnt)
    __builtin_amdgcn_s_wait_asynccnt(0);
#else
    asm volatile("s_wait_asynccnt 0" ::: "memory");
#endif
#else
    *(v8h*)lA       = *(const v8h*)gA;
    *(v8h*)(lA + 8) = *(const v8h*)(gA + 8);
    *(v8h*)lB       = *(const v8h*)gB;
    *(v8h*)(lB + 8) = *(const v8h*)(gB + 8);
#endif
    __syncthreads();

    v16h afrag[2];
#pragma unroll
    for (int i = 0; i < 2; ++i) {
      const _Float16* ap = &As[(waveM * 32 + i * 16 + mrow) * 32 + aKlo];
      v8h lo = *(const v8h*)ap;
      v8h hi = *(const v8h*)(ap + 8);
      afrag[i] = __builtin_shufflevector(lo, hi, 0, 1, 2, 3, 4, 5, 6, 7,
                                         8, 9, 10, 11, 12, 13, 14, 15);
    }
    v16h bfrag[4];
#pragma unroll
    for (int j = 0; j < 4; ++j) {
      const _Float16* bp = &Bs[(waveN * 64 + j * 16 + mrow) * 32 + bKlo];
      v8h lo = *(const v8h*)bp;
      v8h hi = *(const v8h*)(bp + 8);
      bfrag[j] = __builtin_shufflevector(lo, hi, 0, 1, 2, 3, 4, 5, 6, 7,
                                         8, 9, 10, 11, 12, 13, 14, 15);
    }

#pragma unroll
    for (int i = 0; i < 2; ++i)
#pragma unroll
      for (int j = 0; j < 4; ++j)
        acc[i][j] = __builtin_amdgcn_wmma_f32_16x16x32_f16(
            false, afrag[i], false, bfrag[j], (short)0, acc[i][j], false, false);

    __syncthreads();
  }

  // Epilogue: C/D layout: VGPR r -> M = r (lanes 0-15) / 8+r (lanes 16-31); N = lane&15
  const int rOff = (lane >> 4) << 3;  // 0 or 8
#pragma unroll
  for (int i = 0; i < 2; ++i) {
#pragma unroll
    for (int j = 0; j < 4; ++j) {
      int rowBase = blockM + waveM * 32 + i * 16 + rOff;
      int col     = blockN + waveN * 64 + j * 16 + mrow;
      float bv = bias[col];
#pragma unroll
      for (int r = 0; r < 8; ++r)
        C[(size_t)(rowBase + r) * N + col] = acc[i][j][r] + bv;
    }
  }
}

// ---------------------------------------------------------------------------
// Block reduction helper (wave32): intra-wave shfl, then LDS across 8 waves
// ---------------------------------------------------------------------------
__device__ __forceinline__ float blockReduceSum(float val, float* sbuf) {
#pragma unroll
  for (int off = 16; off > 0; off >>= 1) val += __shfl_down(val, off, 32);
  const int lane = threadIdx.x & 31;
  const int wid  = threadIdx.x >> 5;
  if (lane == 0) sbuf[wid] = val;
  __syncthreads();
  if (wid == 0) {
    float v = (lane < 8) ? sbuf[lane] : 0.f;
#pragma unroll
    for (int off = 4; off > 0; off >>= 1) v += __shfl_down(v, off, 32);
    if (lane == 0) sbuf[0] = v;
  }
  __syncthreads();
  float res = sbuf[0];
  __syncthreads();  // sbuf may be reused immediately by caller
  return res;
}

// ---------------------------------------------------------------------------
// Fused LayerNorm + LIF scan. One workgroup per (b,n) row; membrane potential
// lives in registers across the T loop (T is the sequential recurrence dim).
// C = PER * 256 channels. Writes 0/1 spikes as OutT.
// ---------------------------------------------------------------------------
template <int PER, typename OutT>
__global__ __launch_bounds__(256)
void ln_lif_kernel(const float* __restrict__ X, const float* __restrict__ lw,
                   const float* __restrict__ lb, const float* __restrict__ betaPtr,
                   OutT* __restrict__ out, int rows, int T) {
  constexpr int C = PER * 256;
  __shared__ float sbuf[8];
  const int row = blockIdx.x;
  const int tid = threadIdx.x;
  const float beta = betaPtr[0];

  float mem[PER];
#pragma unroll
  for (int i = 0; i < PER; ++i) mem[i] = 0.f;

  for (int t = 0; t < T; ++t) {
    const size_t base = ((size_t)t * rows + row) * C;
    float v[PER];
    float s = 0.f, sq = 0.f;
#pragma unroll
    for (int i = 0; i < PER; ++i) {
      v[i] = X[base + i * 256 + tid];
      s += v[i];
      sq += v[i] * v[i];
    }
    const float tot  = blockReduceSum(s, sbuf);
    const float tot2 = blockReduceSum(sq, sbuf);
    const float mean = tot * (1.f / C);
    const float var  = tot2 * (1.f / C) - mean * mean;
    const float rstd = rsqrtf(var + 1e-5f);
#pragma unroll
    for (int i = 0; i < PER; ++i) {
      const int f = i * 256 + tid;
      const float xn = (v[i] - mean) * rstd * lw[f] + lb[f];
      mem[i] = beta * mem[i] + xn;
      const float spk = (mem[i] > 1.0f) ? 1.0f : 0.0f;  // THRESHOLD = 1.0
      mem[i] -= spk;                                    // soft reset
      out[base + f] = (OutT)spk;
    }
  }
}

// ---------------------------------------------------------------------------
// Host-side orchestration
// ---------------------------------------------------------------------------
extern "C" void kernel_launch(void* const* d_in, const int* in_sizes, int n_in,
                              void* d_out, int out_size, void* d_ws, size_t ws_size,
                              hipStream_t stream) {
  (void)in_sizes; (void)n_in; (void)out_size; (void)ws_size;

  const float* x     = (const float*)d_in[0];
  const float* W1    = (const float*)d_in[1];
  const float* b1    = (const float*)d_in[2];
  const float* ln1w  = (const float*)d_in[3];
  const float* ln1b  = (const float*)d_in[4];
  const float* beta1 = (const float*)d_in[5];
  const float* W2    = (const float*)d_in[6];
  const float* b2    = (const float*)d_in[7];
  const float* ln2w  = (const float*)d_in[8];
  const float* ln2b  = (const float*)d_in[9];
  const float* beta2 = (const float*)d_in[10];
  float* out = (float*)d_out;

  constexpr int T = 4, B = 8, NN = 1024, D = 512, F = 2048;
  constexpr int M    = T * B * NN;  // 32768 GEMM rows
  constexpr int ROWS = B * NN;      // 8192 (b,n) rows

  // Workspace carve-out (deterministic layout)
  char* ws = (char*)d_ws;
  size_t off = 0;
  auto carve = [&](size_t bytes) -> void* {
    void* p = ws + off;
    off += (bytes + 255) & ~(size_t)255;
    return p;
  };
  _Float16* Xh  = (_Float16*)carve((size_t)M * D * sizeof(_Float16));  // x in f16
  _Float16* W1t = (_Float16*)carve((size_t)D * F * sizeof(_Float16));  // W1^T [F][D]
  _Float16* W2t = (_Float16*)carve((size_t)F * D * sizeof(_Float16));  // W2^T [D][F]
  _Float16* S1  = (_Float16*)carve((size_t)M * F * sizeof(_Float16));  // spikes layer1
  float*    H   = (float*)carve((size_t)M * F * sizeof(float));        // GEMM1 out
  float*    Y   = (float*)carve((size_t)M * D * sizeof(float));        // GEMM2 out

  // 1) Precision staging
  cvt_f32_to_f16<<<4096, 256, 0, stream>>>(x, Xh, (size_t)M * D);
  transpose_f32_to_f16<<<2048, 256, 0, stream>>>(W1, W1t, D, F);
  transpose_f32_to_f16<<<2048, 256, 0, stream>>>(W2, W2t, F, D);

  // 2) GEMM1: H = Xh @ W1 + b1   (M x F, K = D)
  gemm_f16_wmma<<<dim3(F / 128, M / 128), 256, 0, stream>>>(Xh, W1t, b1, H, M, F, D);

  // 3) LayerNorm(F) + LIF over T -> f16 spikes
  ln_lif_kernel<8, _Float16><<<ROWS, 256, 0, stream>>>(H, ln1w, ln1b, beta1, S1, ROWS, T);

  // 4) GEMM2: Y = S1 @ W2 + b2   (M x D, K = F)
  gemm_f16_wmma<<<dim3(D / 128, M / 128), 256, 0, stream>>>(S1, W2t, b2, Y, M, D, F);

  // 5) LayerNorm(D) + LIF over T -> f32 spikes into d_out
  ln_lif_kernel<2, float><<<ROWS, 256, 0, stream>>>(Y, ln2w, ln2b, beta2, out, ROWS, T);
}
